// CAM_Module_23828478558746
// MI455X (gfx1250) — compile-verified, matrix operand order checked by reading, not run on hardware
//
#include <hip/hip_runtime.h>
#include <hip/hip_bf16.h>

typedef __attribute__((ext_vector_type(16))) __bf16 v16bf;
typedef __attribute__((ext_vector_type(8)))  float  v8f;
typedef __attribute__((ext_vector_type(4)))  __bf16 v4bf;
typedef __attribute__((ext_vector_type(4)))  float  v4f;

typedef int v4i_vs __attribute__((vector_size(16)));
typedef __attribute__((address_space(1))) v4i_vs* as1_v4i_p;
typedef __attribute__((address_space(3))) v4i_vs* as3_v4i_p;

#define BATCH 16
#define CCH   512
#define NPIX  4096
#define LDP   80   // LDS row pitch in bf16 (32B-aligned rows)

// ---------------------------------------------------------------------------
// CDNA5 async global->LDS (ASYNCcnt-tracked)
// ---------------------------------------------------------------------------
__device__ __forceinline__ void async_b128(const void* g, void* l) {
#if __has_builtin(__builtin_amdgcn_global_load_async_to_lds_b128)
  void* gnc = const_cast<void*>(g);
  __builtin_amdgcn_global_load_async_to_lds_b128(
      (as1_v4i_p)gnc, (as3_v4i_p)l, 0, 0);
#else
  // GV mode: vdst = LDS byte offset (low 32 bits of generic LDS pointer),
  // vaddr = 64-bit global address, no saddr.
  asm volatile("global_load_async_to_lds_b128 %0, %1, off"
               :
               : "v"((unsigned int)(unsigned long long)l), "v"(g)
               : "memory");
#endif
}

#if __has_builtin(__builtin_amdgcn_s_wait_asynccnt)
#define WAIT_ASYNC(n) __builtin_amdgcn_s_wait_asynccnt(n)
#else
#define WAIT_ASYNC(n) asm volatile("s_wait_asynccnt %0" ::"i"(n) : "memory")
#endif

// Issue one 128x64 bf16 tile (rows row0..row0+127, cols k0..k0+63) of a
// row-major matrix with row pitch PITCH into an LDS tile. 256 threads,
// 4 async b128 per thread (8 bf16 each).
template <int PITCH>
__device__ __forceinline__ void issue_tile(const __bf16* __restrict__ g0,
                                           int row0, int k0,
                                           __bf16 (*lds)[LDP], int tid) {
#pragma unroll
  for (int i = 0; i < 4; ++i) {
    int ch = tid + i * 256;          // 0..1023 : 128 rows x 8 chunks
    int r  = ch >> 3;
    int c8 = ch & 7;
    async_b128(g0 + (size_t)(row0 + r) * PITCH + k0 + c8 * 8, &lds[r][c8 * 8]);
  }
}

// ---------------------------------------------------------------------------
// Kernel 0: x (f32, BxCxN) -> v_bf (bf16, BxCxN) and vT (bf16, BxNxC)
// ---------------------------------------------------------------------------
__global__ __launch_bounds__(256) void k_convert(const float* __restrict__ x,
                                                 __bf16* __restrict__ vbf,
                                                 __bf16* __restrict__ vT) {
  __shared__ __bf16 sT[64][68];
  const int b  = blockIdx.z;
  const int c0 = blockIdx.y * 64;
  const int n0 = blockIdx.x * 64;
  const int tid = threadIdx.x;

  const float* src = x + ((size_t)(b * CCH + c0)) * NPIX + n0;
  __bf16* vdst = vbf + ((size_t)(b * CCH + c0)) * NPIX + n0;
#pragma unroll
  for (int i = 0; i < 4; ++i) {
    int ch = tid + i * 256;        // 0..1023 chunks of 4 floats
    int r  = ch >> 4;              // c-local row 0..63
    int c4 = ch & 15;              // n-local chunk
    v4f v  = *(const v4f*)(src + (size_t)r * NPIX + c4 * 4);
    v4bf bv = {(__bf16)v.x, (__bf16)v.y, (__bf16)v.z, (__bf16)v.w};
    *(v4bf*)&sT[r][c4 * 4] = bv;
    *(v4bf*)(vdst + (size_t)r * NPIX + c4 * 4) = bv;   // row-major bf16 copy
  }
  __syncthreads();
  __bf16* dst = vT + ((size_t)(b * NPIX + n0)) * CCH + c0;
#pragma unroll
  for (int i = 0; i < 4; ++i) {
    int ch = tid + i * 256;
    int r2 = ch >> 4;              // n-local row 0..63
    int c4 = ch & 15;              // c-local chunk
    v4bf o = {sT[c4 * 4 + 0][r2], sT[c4 * 4 + 1][r2],
              sT[c4 * 4 + 2][r2], sT[c4 * 4 + 3][r2]};
    *(v4bf*)(dst + (size_t)r2 * CCH + c4 * 4) = o;
  }
}

// ---------------------------------------------------------------------------
// Kernel 1: energy[b] = v_b * v_b^T (512x512, K=4096), bf16 WMMA,
// async double-buffered LDS. grid (4,4,16), 256 threads = 8 waves.
// ---------------------------------------------------------------------------
__global__ __launch_bounds__(256) void k_gram(const __bf16* __restrict__ vbf,
                                              float* __restrict__ energy) {
  __shared__ __bf16 sA[2][128][LDP];
  __shared__ __bf16 sB[2][128][LDP];
  const int b       = blockIdx.z;
  const int rowBase = blockIdx.y * 128;
  const int colBase = blockIdx.x * 128;
  const int tid   = threadIdx.x;
  const int wid   = tid >> 5;
  const int lane  = tid & 31;
  const int lmod  = lane & 15;
  const int lhalf = lane >> 4;
  const int waveM = wid >> 1;   // 0..3 -> 32 rows
  const int waveN = wid & 1;    // 0..1 -> 64 cols

  const __bf16* vb = vbf + (size_t)b * CCH * NPIX;

  const v8f vzero = {0.f, 0.f, 0.f, 0.f, 0.f, 0.f, 0.f, 0.f};
  v8f acc[2][4];
#pragma unroll
  for (int mt = 0; mt < 2; ++mt)
#pragma unroll
    for (int nt = 0; nt < 4; ++nt) acc[mt][nt] = vzero;

  // prologue: fill buffer 0
  issue_tile<NPIX>(vb, rowBase, 0, sA[0], tid);
  issue_tile<NPIX>(vb, colBase, 0, sB[0], tid);

  const int nk = NPIX / 64;
  for (int ki = 0; ki < nk; ++ki) {
    const int buf = ki & 1;
    if (ki + 1 < nk) {
      issue_tile<NPIX>(vb, rowBase, (ki + 1) * 64, sA[buf ^ 1], tid);
      issue_tile<NPIX>(vb, colBase, (ki + 1) * 64, sB[buf ^ 1], tid);
      WAIT_ASYNC(8);   // previous batch of 8 complete (in-order)
    } else {
      WAIT_ASYNC(0);
    }
    __syncthreads();
#pragma unroll
    for (int kk = 0; kk < 64; kk += 32) {
      v16bf af[2], bf[4];
#pragma unroll
      for (int mt = 0; mt < 2; ++mt)
        af[mt] = *(const v16bf*)&sA[buf][waveM * 32 + mt * 16 + lmod][kk + lhalf * 16];
#pragma unroll
      for (int nt = 0; nt < 4; ++nt)
        bf[nt] = *(const v16bf*)&sB[buf][waveN * 64 + nt * 16 + lmod][kk + lhalf * 16];
#pragma unroll
      for (int mt = 0; mt < 2; ++mt)
#pragma unroll
        for (int nt = 0; nt < 4; ++nt)
          acc[mt][nt] = __builtin_amdgcn_wmma_f32_16x16x32_bf16(
              false, af[mt], false, bf[nt], (short)0, acc[mt][nt], false, false);
    }
    __syncthreads();
  }

  float* eb = energy + (size_t)b * CCH * CCH;
#pragma unroll
  for (int mt = 0; mt < 2; ++mt)
#pragma unroll
    for (int nt = 0; nt < 4; ++nt) {
      int col = colBase + waveN * 64 + nt * 16 + lmod;
#pragma unroll
      for (int r = 0; r < 8; ++r) {
        int row = rowBase + waveM * 32 + mt * 16 + lhalf * 8 + r;
        eb[(size_t)row * CCH + col] = acc[mt][nt][r];
      }
    }
}

// ---------------------------------------------------------------------------
// Kernel 2: att[c,d] = exp(rowmin[c]-energy[c,d]) / sum_d exp(...) (bf16 out)
// ---------------------------------------------------------------------------
__global__ __launch_bounds__(256) void k_softmax(const float* __restrict__ energy,
                                                 __bf16* __restrict__ att) {
  __shared__ float red[256];
  const int row = blockIdx.x;                 // b*512 + c
  const int t   = threadIdx.x;
  const float* e = energy + (size_t)row * CCH;

  float v0 = e[t], v1 = e[t + 256];
  red[t] = fminf(v0, v1);
  __syncthreads();
#pragma unroll
  for (int s = 128; s > 0; s >>= 1) {
    if (t < s) red[t] = fminf(red[t], red[t + s]);
    __syncthreads();
  }
  float rmin = red[0];
  __syncthreads();

  float e0 = __expf(rmin - v0);
  float e1 = __expf(rmin - v1);
  red[t] = e0 + e1;
  __syncthreads();
#pragma unroll
  for (int s = 128; s > 0; s >>= 1) {
    if (t < s) red[t] += red[t + s];
    __syncthreads();
  }
  float inv = 1.0f / red[0];

  __bf16* a = att + (size_t)row * CCH;
  a[t]       = (__bf16)(e0 * inv);
  a[t + 256] = (__bf16)(e1 * inv);
}

// ---------------------------------------------------------------------------
// Kernel 3: out = gamma * (att @ v) + x ; bf16 WMMA (K=512), async LDS,
// fused epilogue. grid (32,4,16), 256 threads = 8 waves.
// ---------------------------------------------------------------------------
__global__ __launch_bounds__(256) void k_attn_out(const __bf16* __restrict__ att,
                                                  const __bf16* __restrict__ vT,
                                                  const float* __restrict__ x,
                                                  const float* __restrict__ gamma,
                                                  float* __restrict__ out) {
  __shared__ __bf16 sA[2][128][LDP];
  __shared__ __bf16 sB[2][128][LDP];
  const int b       = blockIdx.z;
  const int rowBase = blockIdx.y * 128;    // C rows
  const int colBase = blockIdx.x * 128;    // N cols
  const int tid   = threadIdx.x;
  const int wid   = tid >> 5;
  const int lane  = tid & 31;
  const int lmod  = lane & 15;
  const int lhalf = lane >> 4;
  const int waveM = wid >> 1;
  const int waveN = wid & 1;

  const __bf16* ab = att + (size_t)b * CCH * CCH;
  const __bf16* tb = vT + (size_t)b * NPIX * CCH;

  const v8f vzero = {0.f, 0.f, 0.f, 0.f, 0.f, 0.f, 0.f, 0.f};
  v8f acc[2][4];
#pragma unroll
  for (int mt = 0; mt < 2; ++mt)
#pragma unroll
    for (int nt = 0; nt < 4; ++nt) acc[mt][nt] = vzero;

  issue_tile<CCH>(ab, rowBase, 0, sA[0], tid);
  issue_tile<CCH>(tb, colBase, 0, sB[0], tid);

  const int nk = CCH / 64;
  for (int ki = 0; ki < nk; ++ki) {
    const int buf = ki & 1;
    if (ki + 1 < nk) {
      issue_tile<CCH>(ab, rowBase, (ki + 1) * 64, sA[buf ^ 1], tid);
      issue_tile<CCH>(tb, colBase, (ki + 1) * 64, sB[buf ^ 1], tid);
      WAIT_ASYNC(8);
    } else {
      WAIT_ASYNC(0);
    }
    __syncthreads();
#pragma unroll
    for (int kk = 0; kk < 64; kk += 32) {
      v16bf af[2], bf[4];
#pragma unroll
      for (int mt = 0; mt < 2; ++mt)
        af[mt] = *(const v16bf*)&sA[buf][waveM * 32 + mt * 16 + lmod][kk + lhalf * 16];
#pragma unroll
      for (int nt = 0; nt < 4; ++nt)
        bf[nt] = *(const v16bf*)&sB[buf][waveN * 64 + nt * 16 + lmod][kk + lhalf * 16];
#pragma unroll
      for (int mt = 0; mt < 2; ++mt)
#pragma unroll
        for (int nt = 0; nt < 4; ++nt)
          acc[mt][nt] = __builtin_amdgcn_wmma_f32_16x16x32_bf16(
              false, af[mt], false, bf[nt], (short)0, acc[mt][nt], false, false);
    }
    __syncthreads();
  }

  const float g = gamma[0];
#pragma unroll
  for (int mt = 0; mt < 2; ++mt)
#pragma unroll
    for (int nt = 0; nt < 4; ++nt) {
      int n = colBase + waveN * 64 + nt * 16 + lmod;
#pragma unroll
      for (int r = 0; r < 8; ++r) {
        int c = rowBase + waveM * 32 + mt * 16 + lhalf * 8 + r;
        size_t idx = (size_t)(b * CCH + c) * NPIX + n;
        out[idx] = g * acc[mt][nt][r] + x[idx];
      }
    }
}

// ---------------------------------------------------------------------------
extern "C" void kernel_launch(void* const* d_in, const int* in_sizes, int n_in,
                              void* d_out, int out_size, void* d_ws, size_t ws_size,
                              hipStream_t stream) {
  const float* x     = (const float*)d_in[0];
  const float* gamma = (const float*)d_in[1];
  float* out = (float*)d_out;

  char* ws = (char*)d_ws;
  // ws layout: v_bf (bf16, 64 MiB) | vT (bf16, 64 MiB) | energy (f32, 16 MiB)
  //          | att (bf16, 8 MiB)
  const size_t vbytes = (size_t)BATCH * CCH * NPIX * sizeof(__bf16);
  __bf16* vbf    = (__bf16*)ws;
  __bf16* vT     = (__bf16*)(ws + vbytes);
  float*  energy = (float*)(ws + 2 * vbytes);
  __bf16* att    = (__bf16*)(ws + 2 * vbytes + (size_t)BATCH * CCH * CCH * sizeof(float));

  k_convert<<<dim3(NPIX / 64, CCH / 64, BATCH), 256, 0, stream>>>(x, vbf, vT);
  k_gram<<<dim3(CCH / 128, CCH / 128, BATCH), 256, 0, stream>>>(vbf, energy);
  k_softmax<<<dim3(BATCH * CCH), 256, 0, stream>>>(energy, att);
  k_attn_out<<<dim3(NPIX / 128, CCH / 128, BATCH), 256, 0, stream>>>(att, vT, x, gamma, out);
}
